// CachedGptOssAttention_45165876085370
// MI455X (gfx1250) — matrix operand order repaired and checked by reading, not compiled, not run on hardware
//
#include <hip/hip_runtime.h>

// ---------------- problem constants ----------------
#define B_   2
#define S_   2048
#define M_   4096
#define NQ_  32
#define NKV_ 8
#define HD_  128
#define WIN_ 1024
#define TOK_ (B_ * S_)           // 4096 tokens
#define QN_  (NQ_ * HD_)         // 4096
#define KVN_ (NKV_ * HD_)        // 1024

typedef __attribute__((ext_vector_type(16))) __bf16 v16bf;
typedef __attribute__((ext_vector_type(8)))  float  v8f;
typedef __attribute__((ext_vector_type(4)))  int    i32x4;

struct U32x8 { uint4 a, b; };

// ---------------- CDNA5 async global->LDS path (guarded) ----------------
#if __has_builtin(__builtin_amdgcn_global_load_async_to_lds_b128)
#define HAVE_ASYNC 1
#else
#define HAVE_ASYNC 0
#endif

#if HAVE_ASYNC
template <int OFF>
__device__ __forceinline__ void async_b128(const void* g, void* l) {
  __builtin_amdgcn_global_load_async_to_lds_b128((i32x4*)g, (i32x4*)l, OFF, 0);
}
#endif

__device__ __forceinline__ void wait_async0() {
#if HAVE_ASYNC
#if __has_builtin(__builtin_amdgcn_s_wait_asynccnt)
  __builtin_amdgcn_s_wait_asynccnt(0);
#else
  asm volatile("s_wait_asynccnt 0x0" ::: "memory");
#endif
#endif
}

__device__ __forceinline__ unsigned short f2bf(float x) {
  unsigned int u = __float_as_uint(x);
  u += 0x7FFFu + ((u >> 16) & 1u);          // round-to-nearest-even
  return (unsigned short)(u >> 16);
}

// Build a 16-element bf16 WMMA operand from two 16-byte chunks.
__device__ __forceinline__ v16bf make16(const unsigned short* p0, const unsigned short* p1) {
  U32x8 t;
  t.a = *(const uint4*)p0;
  t.b = *(const uint4*)p1;
  return __builtin_bit_cast(v16bf, t);
}

__device__ __forceinline__ v8f wmma_bf16(v16bf a, v16bf b, v8f c) {
  return __builtin_amdgcn_wmma_f32_16x16x32_bf16(false, a, false, b, (short)0, c, false, false);
}

// ---------------- elementwise converts ----------------
__global__ void k_f32_to_bf16(const float* __restrict__ in, unsigned short* __restrict__ out, int n) {
  int i = blockIdx.x * blockDim.x + threadIdx.x;
  if (i < n) out[i] = f2bf(in[i]);
}

// wt[n][k] = bf16(w[k][n])
__global__ void k_convert_transpose(const float* __restrict__ w, unsigned short* __restrict__ wt,
                                    int rows, int cols) {
  int i = blockIdx.x * blockDim.x + threadIdx.x;
  if (i >= rows * cols) return;
  int m = i / cols, n = i - m * cols;
  wt[(size_t)n * rows + m] = f2bf(w[i]);
}

// RoPE (half-split) in f32, emit bf16.  x: [B,S,nh,128]
__global__ void k_rope_bf16(const float* __restrict__ xf, unsigned short* __restrict__ xb, int nh) {
  int idx = blockIdx.x * blockDim.x + threadIdx.x;
  int total = B_ * S_ * nh * 64;
  if (idx >= total) return;
  int i = idx & 63;
  int h = (idx >> 6) % nh;
  int s = (idx / (64 * nh)) % S_;
  int b = idx / (64 * nh * S_);
  float inv_freq = __powf(10000.0f, -(float)i * (1.0f / 64.0f));
  float fr = (float)s * inv_freq;
  float cs, sn;
  __sincosf(fr, &sn, &cs);
  size_t base = (((size_t)b * S_ + s) * nh + h) * HD_;
  float x1 = xf[base + i], x2 = xf[base + 64 + i];
  xb[base + i]      = f2bf(x1 * cs - x2 * sn);
  xb[base + 64 + i] = f2bf(x2 * cs + x1 * sn);
}

// V f32 [b][s][kv][hd] -> bf16 Vt [b][kv][hd][s]
__global__ void k_v_transpose(const float* __restrict__ vf, unsigned short* __restrict__ vt) {
  int idx = blockIdx.x * blockDim.x + threadIdx.x;
  if (idx >= B_ * S_ * NKV_ * HD_) return;
  int hd = idx & 127;
  int t  = idx >> 7;
  int kv = t & 7; t >>= 3;
  int s  = t & (S_ - 1);
  int b  = t >> 11;
  vt[(((size_t)b * NKV_ + kv) * HD_ + hd) * S_ + s] = f2bf(vf[idx]);
}

// ---------------- bf16 WMMA GEMM:  C[M,N](f32) = A[M,K](bf16) * Bt[N,K](bf16)^T ----------------
// 128x128 tile, BK=32, 256 threads = 8 waves, wave tile 32x64, double-buffered LDS + async copy.
template <int N, int K>
__global__ __launch_bounds__(256) void k_gemm_bf16(const unsigned short* __restrict__ A,
                                                   const unsigned short* __restrict__ Bt,
                                                   float* __restrict__ C) {
  __shared__ __align__(16) unsigned short As[2][128 * 40];
  __shared__ __align__(16) unsigned short Bs[2][128 * 40];
  const int tid = threadIdx.x, lane = tid & 31, wid = tid >> 5;
  const int wm = wid & 3, wn = wid >> 2;
  const int lr = lane & 15, hh = lane >> 4;
  const int m0 = blockIdx.x * 128, n0 = blockIdx.y * 128;
  const int ldRow = tid >> 1, ldCol = (tid & 1) * 16;

  const unsigned short* ga = A  + (size_t)(m0 + ldRow) * K + ldCol;
  const unsigned short* gb = Bt + (size_t)(n0 + ldRow) * K + ldCol;

  v8f acc[2][4] = {};

  auto stage = [&](int buf, int k0) {
    unsigned short* da = &As[buf][ldRow * 40 + ldCol];
    unsigned short* db = &Bs[buf][ldRow * 40 + ldCol];
#if HAVE_ASYNC
    async_b128<0>(ga + k0, da);
    async_b128<16>(ga + k0, da);
    async_b128<0>(gb + k0, db);
    async_b128<16>(gb + k0, db);
#else
    const uint4* pa = (const uint4*)(ga + k0);
    const uint4* pb = (const uint4*)(gb + k0);
    ((uint4*)da)[0] = pa[0]; ((uint4*)da)[1] = pa[1];
    ((uint4*)db)[0] = pb[0]; ((uint4*)db)[1] = pb[1];
#endif
  };

  stage(0, 0);
  constexpr int KT = K / 32;
#pragma unroll 2
  for (int kt = 0; kt < KT; kt++) {
    const int cur = kt & 1;
    wait_async0();
    __syncthreads();                 // tile `cur` visible; everyone done reading `cur^1`
    if (kt + 1 < KT) stage(cur ^ 1, (kt + 1) * 32);

    const unsigned short* as = As[cur];
    const unsigned short* bs = Bs[cur];
    v16bf a[2], b[4];
#pragma unroll
    for (int i = 0; i < 2; i++) {
      const unsigned short* p = as + (wm * 32 + i * 16 + lr) * 40;
      a[i] = make16(p + 16 * hh, p + 16 * hh + 8);
    }
#pragma unroll
    for (int j = 0; j < 4; j++) {
      const unsigned short* p = bs + (wn * 64 + j * 16 + lr) * 40;
      b[j] = make16(p + 16 * hh, p + 16 * hh + 8);
    }
#pragma unroll
    for (int i = 0; i < 2; i++)
#pragma unroll
      for (int j = 0; j < 4; j++)
        acc[i][j] = wmma_bf16(a[i], b[j], acc[i][j]);
  }

  // epilogue: one base address, compile-time immediate offsets
  float* cp = C + (size_t)(m0 + wm * 32 + 8 * hh) * N + n0 + wn * 64 + lr;
#pragma unroll
  for (int i = 0; i < 2; i++)
#pragma unroll
    for (int j = 0; j < 4; j++)
#pragma unroll
      for (int e = 0; e < 8; e++)
        cp[(size_t)(i * 16 + e) * N + j * 16] = acc[i][j][e];
}

// ---------------- flash attention with sink softmax + sliding window ----------------
// grid: (S/64, NQ, B), block: 128 threads = 4 waves; each wave owns 16 query rows.
// Computes S^T = K*Q^T (softmax over k register-resident), out^T += V^T * P.
// K/V tiles double-buffered in LDS with async global->LDS copies.
__global__ __launch_bounds__(128) void k_attn(const unsigned short* __restrict__ Qb,
                                              const unsigned short* __restrict__ Kb,
                                              const unsigned short* __restrict__ Vt,
                                              const float* __restrict__ sinks,
                                              unsigned short* __restrict__ Ob) {
  __shared__ __align__(16) unsigned short Ks[2][32 * 136];
  __shared__ __align__(16) unsigned short Vs[2][128 * 40];
  __shared__ __align__(16) unsigned short Ps[4 * 16 * 40];

  const int tid = threadIdx.x, lane = tid & 31, wid = tid >> 5;
  const int lr = lane & 15, hh = lane >> 4;
  const int qt = blockIdx.x, head = blockIdx.y, b = blockIdx.z, kv = head >> 2;
  const int qbb = qt * 64;
  const int qrow = qbb + wid * 16 + lr;

  // Q as B-operand: [32(hd) x 16(q)] per 32-wide hd chunk, q = lane
  v16bf qop[4];
  const unsigned short* qp = Qb + (((size_t)b * S_ + qrow) * NQ_ + head) * HD_;
#pragma unroll
  for (int c = 0; c < 4; c++)
    qop[c] = make16(qp + c * 32 + 16 * hh, qp + c * 32 + 16 * hh + 8);

  float m = sinks[head];   // sink logit initializes running max
  float l = 1.0f;          // exp(sink - m) = 1
  v8f acc[8] = {};

  const int kRow = tid >> 2, kCol = (tid & 3) * 32;
  const unsigned short* gK = Kb + (((size_t)b * S_ + kRow) * NKV_ + kv) * HD_ + kCol;
  const unsigned short* gV = Vt + (((size_t)b * NKV_ + kv) * HD_ + tid) * S_;

  auto stage = [&](int buf, int kb) {
    unsigned short* dK = &Ks[buf][kRow * 136 + kCol];
    unsigned short* dV = &Vs[buf][tid * 40];
    const unsigned short* sK = gK + (size_t)kb * (NKV_ * HD_);
    const unsigned short* sV = gV + kb;
#if HAVE_ASYNC
    async_b128<0>(sK, dK);  async_b128<16>(sK, dK);
    async_b128<32>(sK, dK); async_b128<48>(sK, dK);
    async_b128<0>(sV, dV);  async_b128<16>(sV, dV);
    async_b128<32>(sV, dV); async_b128<48>(sV, dV);
#else
    const uint4* pk = (const uint4*)sK; uint4* qk = (uint4*)dK;
    qk[0] = pk[0]; qk[1] = pk[1]; qk[2] = pk[2]; qk[3] = pk[3];
    const uint4* pv = (const uint4*)sV; uint4* qv = (uint4*)dV;
    qv[0] = pv[0]; qv[1] = pv[1]; qv[2] = pv[2]; qv[3] = pv[3];
#endif
  };

  int kmin = qbb - (WIN_ - 1);
  if (kmin < 0) kmin = 0;
  const int kb0 = kmin & ~31;
  const int nIter = (qbb + 64 - kb0) >> 5;

  stage(0, kb0);
  for (int it = 0; it < nIter; it++) {
    const int kb = kb0 + it * 32;
    const int cur = it & 1;
    wait_async0();
    __syncthreads();
    if (it + 1 < nIter) stage(cur ^ 1, kb + 32);
    const unsigned short* ks = Ks[cur];
    const unsigned short* vs = Vs[cur];

    // scores: S^T[k, q] for k = kb..kb+31
    float p[2][8];
    float mloc = -3.0e38f;
#pragma unroll
    for (int t = 0; t < 2; t++) {
      v8f sacc = {};
#pragma unroll
      for (int c = 0; c < 4; c++) {
        const unsigned short* kr = ks + (t * 16 + lr) * 136 + c * 32;
        sacc = wmma_bf16(make16(kr + 16 * hh, kr + 16 * hh + 8), qop[c], sacc);
      }
#pragma unroll
      for (int j = 0; j < 8; j++) {
        int kg = kb + t * 16 + j + 8 * hh;
        float v = sacc[j] * 0.08838834764831845f;      // 1/sqrt(128)
        bool ok = (kg <= qrow) && (qrow - kg < WIN_);
        v = ok ? v : -1.0e30f;
        p[t][j] = v;
        mloc = fmaxf(mloc, v);
      }
    }
    mloc = fmaxf(mloc, __shfl_xor(mloc, 16, 32));
    float mnew = fmaxf(m, mloc);
    float alpha = __expf(m - mnew);
    float ssum = 0.0f;
#pragma unroll
    for (int t = 0; t < 2; t++)
#pragma unroll
      for (int j = 0; j < 8; j++) {
        p[t][j] = __expf(p[t][j] - mnew);
        ssum += p[t][j];
      }
    ssum += __shfl_xor(ssum, 16, 32);
    l = l * alpha + ssum;
    m = mnew;
#pragma unroll
    for (int i = 0; i < 8; i++)
#pragma unroll
      for (int e = 0; e < 8; e++) acc[i][e] *= alpha;

    // re-layout P through per-wave LDS: [16 q][32 k], stride 40
    unsigned short* pw = Ps + wid * (16 * 40);
#pragma unroll
    for (int t = 0; t < 2; t++)
#pragma unroll
      for (int j = 0; j < 8; j++)
        pw[lr * 40 + t * 16 + j + 8 * hh] = f2bf(p[t][j]);
    asm volatile("s_wait_dscnt 0x0" ::: "memory");
    v16bf pop = make16(pw + lr * 40 + 16 * hh, pw + lr * 40 + 16 * hh + 8);

    // out^T[hd, q] += V^T[hd, k] * P[k, q]
#pragma unroll
    for (int i = 0; i < 8; i++) {
      const unsigned short* vr = vs + (i * 16 + lr) * 40;
      acc[i] = wmma_bf16(make16(vr + 16 * hh, vr + 16 * hh + 8), pop, acc[i]);
    }
  }

  float rl = 1.0f / l;   // denominator includes the sink term
  unsigned short* op = Ob + (((size_t)b * S_ + qrow) * NQ_ + head) * HD_;
#pragma unroll
  for (int i = 0; i < 8; i++)
#pragma unroll
    for (int e = 0; e < 8; e += 2) {
      int hd = i * 16 + e + 8 * hh;
      unsigned int lo = f2bf(acc[i][e] * rl);
      unsigned int hi = f2bf(acc[i][e + 1] * rl);
      *(unsigned int*)(op + hd) = lo | (hi << 16);
    }
}

// ---------------- launcher ----------------
extern "C" void kernel_launch(void* const* d_in, const int* in_sizes, int n_in,
                              void* d_out, int out_size, void* d_ws, size_t ws_size,
                              hipStream_t stream) {
  const float* hs    = (const float*)d_in[0];
  // d_in[1] = attention_mask (recomputed analytically, unused)
  const float* Wq    = (const float*)d_in[2];
  const float* Wk    = (const float*)d_in[3];
  const float* Wv    = (const float*)d_in[4];
  const float* Wo    = (const float*)d_in[5];
  const float* sinks = (const float*)d_in[6];
  float* out = (float*)d_out;

  char* ws = (char*)d_ws;
  size_t off = 0;
  auto alloc = [&](size_t bytes) -> void* {
    void* p = ws + off;
    off += (bytes + 255) & ~(size_t)255;
    return p;
  };
  unsigned short* Xbf = (unsigned short*)alloc((size_t)TOK_ * M_ * 2);
  unsigned short* Wqt = (unsigned short*)alloc((size_t)QN_ * M_ * 2);
  unsigned short* Wkt = (unsigned short*)alloc((size_t)KVN_ * M_ * 2);
  unsigned short* Wvt = (unsigned short*)alloc((size_t)KVN_ * M_ * 2);
  unsigned short* Wot = (unsigned short*)alloc((size_t)M_ * QN_ * 2);
  float*          Qf  = (float*)alloc((size_t)TOK_ * QN_ * 4);
  float*          Kf  = (float*)alloc((size_t)TOK_ * KVN_ * 4);
  float*          Vf  = (float*)alloc((size_t)TOK_ * KVN_ * 4);
  unsigned short* Qbf = (unsigned short*)alloc((size_t)TOK_ * QN_ * 2);
  unsigned short* Kbf = (unsigned short*)alloc((size_t)TOK_ * KVN_ * 2);
  unsigned short* Vtb = (unsigned short*)alloc((size_t)TOK_ * KVN_ * 2);
  unsigned short* Ob  = (unsigned short*)alloc((size_t)TOK_ * QN_ * 2);

  const int T = 256;
  // 1) converts
  k_f32_to_bf16<<<(TOK_ * M_ + T - 1) / T, T, 0, stream>>>(hs, Xbf, TOK_ * M_);
  k_convert_transpose<<<(M_ * QN_  + T - 1) / T, T, 0, stream>>>(Wq, Wqt, M_, QN_);
  k_convert_transpose<<<(M_ * KVN_ + T - 1) / T, T, 0, stream>>>(Wk, Wkt, M_, KVN_);
  k_convert_transpose<<<(M_ * KVN_ + T - 1) / T, T, 0, stream>>>(Wv, Wvt, M_, KVN_);
  k_convert_transpose<<<(QN_ * M_  + T - 1) / T, T, 0, stream>>>(Wo, Wot, QN_, M_);

  // 2) QKV projections (WMMA GEMM)
  k_gemm_bf16<QN_,  M_><<<dim3(TOK_ / 128, QN_  / 128), 256, 0, stream>>>(Xbf, Wqt, Qf);
  k_gemm_bf16<KVN_, M_><<<dim3(TOK_ / 128, KVN_ / 128), 256, 0, stream>>>(Xbf, Wkt, Kf);
  k_gemm_bf16<KVN_, M_><<<dim3(TOK_ / 128, KVN_ / 128), 256, 0, stream>>>(Xbf, Wvt, Vf);

  // 3) RoPE (f32) -> bf16 Q/K, V -> transposed bf16
  k_rope_bf16<<<(B_ * S_ * NQ_  * 64 + T - 1) / T, T, 0, stream>>>(Qf, Qbf, NQ_);
  k_rope_bf16<<<(B_ * S_ * NKV_ * 64 + T - 1) / T, T, 0, stream>>>(Kf, Kbf, NKV_);
  k_v_transpose<<<(B_ * S_ * NKV_ * HD_ + T - 1) / T, T, 0, stream>>>(Vf, Vtb);

  // 4) windowed flash attention with sink softmax
  k_attn<<<dim3(S_ / 64, NQ_, B_), 128, 0, stream>>>(Qbf, Kbf, Vtb, sinks, Ob);

  // 5) output projection -> f32 d_out
  k_gemm_bf16<M_, QN_><<<dim3(TOK_ / 128, M_ / 128), 256, 0, stream>>>(Ob, Wot, out);
}
